// CausalSelfAttention_47064251629913
// MI455X (gfx1250) — compile-verified, matrix operand order checked by reading, not run on hardware
//
#include <hip/hip_runtime.h>
#include <hip/hip_bf16.h>
#include <math.h>

// ---------------------------------------------------------------------------
// Causal self-attention forward, bf16 WMMA pipeline for gfx1250 (MI455X).
//   B=2, S=2048, D=1024, H=16, HD=64
// Stages: cvt/transpose -> QKV proj (WMMA, ping-pong K loop) ->
//         flash attention (WMMA + TDM async K/V staging) -> O proj.
// ---------------------------------------------------------------------------

typedef __bf16 bf16;
typedef __attribute__((ext_vector_type(16))) __bf16 v16bf;
typedef __attribute__((ext_vector_type(8)))  __bf16 v8bf;
typedef __attribute__((ext_vector_type(8)))  float  v8f;
typedef __attribute__((ext_vector_type(4)))  unsigned int u32x4;
typedef __attribute__((ext_vector_type(8)))  int i32x8;
typedef __attribute__((ext_vector_type(4)))  int i32x4;

constexpr int Bc  = 2;
constexpr int Sc  = 2048;
constexpr int Dc  = 1024;
constexpr int Hc  = 16;
constexpr int HDc = 64;
constexpr int Mc  = Bc * Sc;        // 4096 rows for the projection GEMMs

#if defined(__has_builtin)
#if __has_builtin(__builtin_amdgcn_tensor_load_to_lds) && \
    __has_builtin(__builtin_amdgcn_s_wait_tensorcnt)
#define USE_TDM 1
#endif
#endif
#ifndef USE_TDM
#define USE_TDM 0
#endif

// D = A(16x32 bf16) * B(32x16 bf16) + C(16x16 f32)
__device__ inline v8f wmma_bf16(v16bf a, v16bf b, v8f c) {
  return __builtin_amdgcn_wmma_f32_16x16x32_bf16(
      /*neg_a=*/false, a, /*neg_b=*/false, b,
      /*c_mod=*/(short)0, c, /*reuse_a=*/false, /*reuse_b=*/false);
}

// Load a 16x32 bf16 A-fragment (or mirrored B-fragment) from a row-major
// tile with leading dimension `ld` (elements).
// ISA layout (CDNA5 ISA):
//   lanes 0-15 : row m=lane,    K = {0..7, 16..23}
//   lanes 16-31: row m=lane-16, K = {8..15, 24..31}
// -> per lane: two contiguous 16-byte chunks.
__device__ inline v16bf load_frag(const bf16* base, int ld) {
  const int lane = threadIdx.x & 31;
  const int m    = lane & 15;
  const int kh   = lane >> 4;              // 0 or 1
  const bf16* p  = base + m * ld + kh * 8;
  v8bf lo = *(const v8bf*)(p);
  v8bf hi = *(const v8bf*)(p + 16);
  v16bf f;
#pragma unroll
  for (int i = 0; i < 8; ++i) { f[i] = lo[i]; f[i + 8] = hi[i]; }
  return f;
}

#if USE_TDM
// Issue a 2D TDM tile load: tile (tile0 x tile1) of bf16 elements, row stride
// `stride0` elements, from global `g` into LDS byte-address `lds_addr`.
// Descriptor bit layout per CDNA5 ISA.
// This toolchain's builtin takes 6 args: (g0, g1, g2, g3, g_extra, cpol).
__device__ inline void tdm_load_2d(unsigned lds_addr, const bf16* g,
                                   unsigned tile0, unsigned tile1,
                                   unsigned long long stride0,
                                   unsigned td0, unsigned td1) {
  unsigned long long ga = (unsigned long long)(uintptr_t)g;
  u32x4 g0;
  g0[0] = 1u;                                       // count=1, user mode
  g0[1] = lds_addr;                                 // lds_addr [63:32]
  g0[2] = (unsigned)ga;                             // global_addr [95:64]
  g0[3] = (unsigned)((ga >> 32) & 0x1FFFFFFu)       // global_addr [120:96]
        | (2u << 30);                               // type=2 ("image")
  i32x8 g1;
  g1[0] = (int)(1u << 16);                          // data_size=1 (2 bytes)
  g1[1] = (int)((td0 & 0xFFFFu) << 16);             // tensor_dim0 lo16
  g1[2] = (int)(((td0 >> 16) & 0xFFFFu) | ((td1 & 0xFFFFu) << 16));
  g1[3] = (int)(((td1 >> 16) & 0xFFFFu) | ((tile0 & 0xFFFFu) << 16));
  g1[4] = (int)(tile1 & 0xFFFFu);                   // tile_dim1; tile_dim2=0
  g1[5] = (int)(stride0 & 0xFFFFFFFFull);           // tensor_dim0_stride lo
  g1[6] = (int)((stride0 >> 32) & 0xFFFFull);       // stride hi; dim1_stride=0
  g1[7] = 0;
  i32x4 z4 = {};
  i32x8 z8 = {};
  __builtin_amdgcn_tensor_load_to_lds(g0, g1, z4, z4, z8, 0);
}
__device__ inline unsigned lds_byte_addr(const void* p) {
  return (unsigned)(uintptr_t)p;                    // LDS offset = addr[31:0]
}
#endif

// ---------------------------------------------------------------------------
// Stage 1: fp32 -> bf16 conversion / transpose
// ---------------------------------------------------------------------------
__global__ void cvt_bf16_kernel(const float* __restrict__ src,
                                bf16* __restrict__ dst, int n) {
  int i = blockIdx.x * blockDim.x + threadIdx.x;
  int stride = gridDim.x * blockDim.x;
  for (; i < n; i += stride) dst[i] = (bf16)src[i];
}

// wt[n*D + k] = w[k*D + n]  (store W^T so B-fragments are contiguous rows)
__global__ void transpose_bf16_kernel(const float* __restrict__ w,
                                      bf16* __restrict__ wt) {
  int i = blockIdx.x * blockDim.x + threadIdx.x;
  int stride = gridDim.x * blockDim.x;
  for (; i < Dc * Dc; i += stride) {
    int n = i >> 10;
    int k = i & (Dc - 1);
    wt[i] = (bf16)w[k * Dc + n];
  }
}

// ---------------------------------------------------------------------------
// Stage 2: QKV projections.  One wave -> one 32x64 strip (2 M-tiles reuse
// each B fragment).  K loop unrolled x2 as an A/B register ping-pong: each
// fragment set loads into its own registers and is consumed directly, so
// there is no rotation-mov traffic between iterations.
// grid = (Mc/32/4, Dc/64, 3), block = 128 (4 waves).
// Q,K written [B,H,S,HD]; V written transposed [B,H,HD,S].
// ---------------------------------------------------------------------------
__global__ void __launch_bounds__(128)
qkv_proj_kernel(const bf16* __restrict__ xb,
                const bf16* __restrict__ wtq,
                const bf16* __restrict__ wtk,
                const bf16* __restrict__ wtv,
                bf16* __restrict__ qo,
                bf16* __restrict__ ko,
                bf16* __restrict__ vto) {
  const int wave = threadIdx.x >> 5;
  const int lane = threadIdx.x & 31;
  const int mt   = blockIdx.x * 4 + wave;    // 32-row tile index, 0..127
  const int ng   = blockIdx.y;               // 64-col group, 0..15
  const int which = blockIdx.z;              // 0:Q 1:K 2:V
  const bf16* wt = (which == 0) ? wtq : (which == 1) ? wtk : wtv;

  v8f acc[2][4];
#pragma unroll
  for (int t = 0; t < 2; ++t)
#pragma unroll
    for (int j = 0; j < 4; ++j) acc[t][j] = {};

  const bf16* arow0 = xb + (size_t)mt * 32 * Dc;
  const bf16* arow1 = arow0 + 16 * Dc;
  const bf16* brow  = wt + (size_t)ng * 64 * Dc;

  // phase-A fragments for kk = 0
  v16bf a0A = load_frag(arow0, Dc);
  v16bf a1A = load_frag(arow1, Dc);
  v16bf bA[4];
#pragma unroll
  for (int j = 0; j < 4; ++j) bA[j] = load_frag(brow + j * 16 * Dc, Dc);

  for (int kk = 0; kk < Dc; kk += 64) {      // Dc % 64 == 0 -> no tail
    // load phase-B fragments (kk+32) while phase-A WMMAs issue
    v16bf a0B = load_frag(arow0 + kk + 32, Dc);
    v16bf a1B = load_frag(arow1 + kk + 32, Dc);
    v16bf bB[4];
#pragma unroll
    for (int j = 0; j < 4; ++j) bB[j] = load_frag(brow + j * 16 * Dc + kk + 32, Dc);
    __builtin_prefetch(arow0 + kk + 64, 0, 1);   // global_prefetch_b8

#pragma unroll
    for (int j = 0; j < 4; ++j) acc[0][j] = wmma_bf16(a0A, bA[j], acc[0][j]);
#pragma unroll
    for (int j = 0; j < 4; ++j) acc[1][j] = wmma_bf16(a1A, bA[j], acc[1][j]);

    // load next phase-A fragments (kk+64) while phase-B WMMAs issue
    if (kk + 64 < Dc) {
      a0A = load_frag(arow0 + kk + 64, Dc);
      a1A = load_frag(arow1 + kk + 64, Dc);
#pragma unroll
      for (int j = 0; j < 4; ++j) bA[j] = load_frag(brow + j * 16 * Dc + kk + 64, Dc);
    }
#pragma unroll
    for (int j = 0; j < 4; ++j) acc[0][j] = wmma_bf16(a0B, bB[j], acc[0][j]);
#pragma unroll
    for (int j = 0; j < 4; ++j) acc[1][j] = wmma_bf16(a1B, bB[j], acc[1][j]);
  }

  const int half = lane >> 4;
  const int nl   = lane & 15;
#pragma unroll
  for (int t = 0; t < 2; ++t) {
#pragma unroll
    for (int j = 0; j < 4; ++j) {
#pragma unroll
      for (int r = 0; r < 8; ++r) {
        int R  = mt * 32 + t * 16 + r + 8 * half;  // global row in [0, 4096)
        int bb = R >> 11;
        int s  = R & (Sc - 1);
        int n  = ng * 64 + j * 16 + nl;
        int h  = n >> 6;
        int hd = n & 63;
        bf16 val = (bf16)acc[t][j][r];
        if (which == 0)
          qo[(((size_t)bb * Hc + h) * Sc + s) * HDc + hd] = val;
        else if (which == 1)
          ko[(((size_t)bb * Hc + h) * Sc + s) * HDc + hd] = val;
        else
          vto[(((size_t)bb * Hc + h) * HDc + hd) * Sc + s] = val;
      }
    }
  }
}

// ---------------------------------------------------------------------------
// Stage 3: causal flash attention.  One wave per 16-query tile, 32 keys/iter.
// K/V tiles staged in per-wave LDS double buffers by the Tensor Data Mover
// (prefetch block i+1 while computing block i), synced with s_wait_tensorcnt.
// grid = (Sc/16/4, Hc, Bc), block = 128.
// ---------------------------------------------------------------------------
__global__ void __launch_bounds__(128)
flash_attn_kernel(const bf16* __restrict__ qb,
                  const bf16* __restrict__ kbuf,
                  const bf16* __restrict__ vtb,
                  bf16* __restrict__ ob) {
  __shared__ __align__(16) bf16 psh[4][16 * 32];   // per-wave P re-layout tile
#if USE_TDM
  __shared__ __align__(16) bf16 ktile[4][2][32 * 64]; // keys x hd
  __shared__ __align__(16) bf16 vtile[4][2][64 * 32]; // hd x keys
#endif
  const int wave = threadIdx.x >> 5;
  const int lane = threadIdx.x & 31;
  const int qt   = blockIdx.x * 4 + wave;          // 0..127
  const int h    = blockIdx.y;
  const int b    = blockIdx.z;
  const int bh   = b * Hc + h;
  const bf16* qp = qb   + (size_t)bh * Sc * HDc;   // [S, 64]
  const bf16* kp = kbuf + (size_t)bh * Sc * HDc;   // [S, 64]
  const bf16* vp = vtb  + (size_t)bh * HDc * Sc;   // [64, S] (V^T)
  const int qbase = qt * 16;
  const int half  = lane >> 4;
  const int nl    = lane & 15;

  v16bf q0 = load_frag(qp + (size_t)qbase * HDc, HDc);       // hd 0..31
  v16bf q1 = load_frag(qp + (size_t)qbase * HDc + 32, HDc);  // hd 32..63

  v8f acc[4];
#pragma unroll
  for (int j = 0; j < 4; ++j) acc[j] = {};
  float mi[8], li[8];
#pragma unroll
  for (int r = 0; r < 8; ++r) { mi[r] = -INFINITY; li[r] = 0.f; }

  bf16* ps = psh[wave];
  const float sc = 0.125f;                         // 1/sqrt(64)
  const int nblocks = (qbase + 15) / 32 + 1;       // causal kv-block count

#if USE_TDM
  // prime the pipeline: block 0 into buffer 0
  tdm_load_2d(lds_byte_addr(&ktile[wave][0][0]), kp, 64, 32, HDc, HDc, Sc);
  tdm_load_2d(lds_byte_addr(&vtile[wave][0][0]), vp, 32, 64, Sc, Sc, HDc);
#endif

  for (int ib = 0; ib < nblocks; ++ib) {
    const int kbase = ib * 32;
#if USE_TDM
    const int cur = ib & 1;
    if (ib + 1 < nblocks) {                        // prefetch next block
      const int nxt = cur ^ 1;
      const int kn  = kbase + 32;
      tdm_load_2d(lds_byte_addr(&ktile[wave][nxt][0]), kp + (size_t)kn * HDc,
                  64, 32, HDc, HDc, Sc);
      tdm_load_2d(lds_byte_addr(&vtile[wave][nxt][0]), vp + kn,
                  32, 64, Sc, Sc, HDc);
      __builtin_amdgcn_s_wait_tensorcnt(2);        // current 2 ops complete
    } else {
      __builtin_amdgcn_s_wait_tensorcnt(0);
    }
    asm volatile("" ::: "memory");
    const bf16* kt = &ktile[wave][cur][0];         // [32, 64]
    const bf16* vt = &vtile[wave][cur][0];         // [64, 32]
#else
    const bf16* kt = kp + (size_t)kbase * HDc;     // [32, 64] in global
    const bf16* vt = nullptr;
#endif

    // scores: two 16x16 tiles covering keys [kbase, kbase+32)
    v8f s0 = {}, s1 = {};
    s0 = wmma_bf16(q0, load_frag(kt, HDc), s0);
    s0 = wmma_bf16(q1, load_frag(kt + 32, HDc), s0);
    s1 = wmma_bf16(q0, load_frag(kt + 16 * HDc, HDc), s1);
    s1 = wmma_bf16(q1, load_frag(kt + 16 * HDc + 32, HDc), s1);

    const bool needmask = (kbase + 31 > qbase);    // wave-uniform
#pragma unroll
    for (int r = 0; r < 8; ++r) {
      float v0 = s0[r] * sc, v1 = s1[r] * sc;
      if (needmask) {
        int qi = qbase + r + 8 * half;
        if (kbase + nl      > qi) v0 = -INFINITY;
        if (kbase + 16 + nl > qi) v1 = -INFINITY;
      }
      s0[r] = v0; s1[r] = v1;
    }

    // online softmax: row stats live across the 16 lanes of each half-wave
    float alpha[8];
#pragma unroll
    for (int r = 0; r < 8; ++r) {
      float rm = fmaxf(s0[r], s1[r]);
#pragma unroll
      for (int off = 1; off < 16; off <<= 1)
        rm = fmaxf(rm, __shfl_xor(rm, off, 32));
      float mnew = fmaxf(mi[r], rm);
      float a  = __expf(mi[r] - mnew);
      float p0 = __expf(s0[r] - mnew);
      float p1 = __expf(s1[r] - mnew);
      float rs = p0 + p1;
#pragma unroll
      for (int off = 1; off < 16; off <<= 1)
        rs += __shfl_xor(rs, off, 32);
      li[r] = li[r] * a + rs;
      mi[r] = mnew;
      alpha[r] = a;
      // park P (C-layout) in LDS row-major so it can be re-read as A-layout
      ps[(r + 8 * half) * 32 + nl]      = (bf16)p0;
      ps[(r + 8 * half) * 32 + 16 + nl] = (bf16)p1;
    }
#pragma unroll
    for (int j = 0; j < 4; ++j) {
#pragma unroll
      for (int r = 0; r < 8; ++r) acc[j][r] *= alpha[r];
    }

    v16bf pf = load_frag(ps, 32);                  // P as 16x32 A-fragment
#pragma unroll
    for (int j = 0; j < 4; ++j) {
#if USE_TDM
      v16bf vf = load_frag(vt + j * 16 * 32, 32);
#else
      v16bf vf = load_frag(vp + (size_t)(j * 16) * Sc + kbase, Sc);
#endif
      acc[j] = wmma_bf16(pf, vf, acc[j]);
    }
  }

  // epilogue: normalize and write merged-head bf16 activations [B,S,D]
#pragma unroll
  for (int j = 0; j < 4; ++j) {
#pragma unroll
    for (int r = 0; r < 8; ++r) {
      int s = qbase + r + 8 * half;
      float o = acc[j][r] / li[r];
      ob[((size_t)b * Sc + s) * Dc + h * 64 + j * 16 + nl] = (bf16)o;
    }
  }
}

// ---------------------------------------------------------------------------
// Stage 4: output projection -> fp32 d_out.  32x64 per wave, ping-pong K loop.
// grid = (Mc/32/4, Dc/64), block = 128.
// ---------------------------------------------------------------------------
__global__ void __launch_bounds__(128)
out_proj_kernel(const bf16* __restrict__ ab,
                const bf16* __restrict__ wto,
                float* __restrict__ out) {
  const int wave = threadIdx.x >> 5;
  const int lane = threadIdx.x & 31;
  const int mt   = blockIdx.x * 4 + wave;
  const int ng   = blockIdx.y;

  v8f acc[2][4];
#pragma unroll
  for (int t = 0; t < 2; ++t)
#pragma unroll
    for (int j = 0; j < 4; ++j) acc[t][j] = {};

  const bf16* arow0 = ab + (size_t)mt * 32 * Dc;
  const bf16* arow1 = arow0 + 16 * Dc;
  const bf16* brow  = wto + (size_t)ng * 64 * Dc;

  v16bf a0A = load_frag(arow0, Dc);
  v16bf a1A = load_frag(arow1, Dc);
  v16bf bA[4];
#pragma unroll
  for (int j = 0; j < 4; ++j) bA[j] = load_frag(brow + j * 16 * Dc, Dc);

  for (int kk = 0; kk < Dc; kk += 64) {
    v16bf a0B = load_frag(arow0 + kk + 32, Dc);
    v16bf a1B = load_frag(arow1 + kk + 32, Dc);
    v16bf bB[4];
#pragma unroll
    for (int j = 0; j < 4; ++j) bB[j] = load_frag(brow + j * 16 * Dc + kk + 32, Dc);
    __builtin_prefetch(arow0 + kk + 64, 0, 1);

#pragma unroll
    for (int j = 0; j < 4; ++j) acc[0][j] = wmma_bf16(a0A, bA[j], acc[0][j]);
#pragma unroll
    for (int j = 0; j < 4; ++j) acc[1][j] = wmma_bf16(a1A, bA[j], acc[1][j]);

    if (kk + 64 < Dc) {
      a0A = load_frag(arow0 + kk + 64, Dc);
      a1A = load_frag(arow1 + kk + 64, Dc);
#pragma unroll
      for (int j = 0; j < 4; ++j) bA[j] = load_frag(brow + j * 16 * Dc + kk + 64, Dc);
    }
#pragma unroll
    for (int j = 0; j < 4; ++j) acc[0][j] = wmma_bf16(a0B, bB[j], acc[0][j]);
#pragma unroll
    for (int j = 0; j < 4; ++j) acc[1][j] = wmma_bf16(a1B, bB[j], acc[1][j]);
  }

  const int half = lane >> 4;
  const int nl   = lane & 15;
#pragma unroll
  for (int t = 0; t < 2; ++t) {
#pragma unroll
    for (int j = 0; j < 4; ++j) {
#pragma unroll
      for (int r = 0; r < 8; ++r) {
        int R = mt * 32 + t * 16 + r + 8 * half;
        int n = ng * 64 + j * 16 + nl;
        out[(size_t)R * Dc + n] = acc[t][j][r];
      }
    }
  }
}

// ---------------------------------------------------------------------------
// Host-side launch
// ---------------------------------------------------------------------------
extern "C" void kernel_launch(void* const* d_in, const int* in_sizes, int n_in,
                              void* d_out, int out_size, void* d_ws, size_t ws_size,
                              hipStream_t stream) {
  const float* x  = (const float*)d_in[0];
  const float* Wq = (const float*)d_in[1];
  const float* Wk = (const float*)d_in[2];
  const float* Wv = (const float*)d_in[3];
  const float* Wo = (const float*)d_in[4];
  float* out = (float*)d_out;

  char* ws = (char*)d_ws;
  const size_t MB = 1u << 20;
  bf16* xb   = (bf16*)(ws + 0 * MB);    // [4096,1024]           8 MiB
  bf16* wtq  = (bf16*)(ws + 8 * MB);    // [1024,1024] = Wq^T    2 MiB
  bf16* wtk  = (bf16*)(ws + 10 * MB);
  bf16* wtv  = (bf16*)(ws + 12 * MB);
  bf16* wto  = (bf16*)(ws + 14 * MB);
  bf16* qbuf = (bf16*)(ws + 16 * MB);   // [B,H,S,HD]            8 MiB
  bf16* kbuf = (bf16*)(ws + 24 * MB);   // [B,H,S,HD]            8 MiB
  bf16* vtbf = (bf16*)(ws + 32 * MB);   // [B,H,HD,S] (V^T)      8 MiB
  bf16* attn = (bf16*)(ws + 40 * MB);   // [B,S,D]               8 MiB
  // total 48 MiB

  // Stage 1
  cvt_bf16_kernel<<<2048, 256, 0, stream>>>(x, xb, Mc * Dc);
  transpose_bf16_kernel<<<1024, 256, 0, stream>>>(Wq, wtq);
  transpose_bf16_kernel<<<1024, 256, 0, stream>>>(Wk, wtk);
  transpose_bf16_kernel<<<1024, 256, 0, stream>>>(Wv, wtv);
  transpose_bf16_kernel<<<1024, 256, 0, stream>>>(Wo, wto);

  // Stage 2: QKV projections (32x64 per wave)
  qkv_proj_kernel<<<dim3(Mc / 32 / 4, Dc / 64, 3), 128, 0, stream>>>(
      xb, wtq, wtk, wtv, qbuf, kbuf, vtbf);

  // Stage 3: fused causal attention with TDM-staged K/V
  flash_attn_kernel<<<dim3(Sc / 16 / 4, Hc, Bc), 128, 0, stream>>>(
      qbuf, kbuf, vtbf, attn);

  // Stage 4: output projection (fp32 result)
  out_proj_kernel<<<dim3(Mc / 32 / 4, Dc / 64), 128, 0, stream>>>(
      attn, wto, out);

  (void)in_sizes; (void)n_in; (void)out_size; (void)ws_size;
}
